// InEKFLevel15_DARE_51170240365113
// MI455X (gfx1250) — matrix operand order repaired and checked by reading, not compiled, not run on hardware
//
#include <hip/hip_runtime.h>
#include <math.h>

// ---------------- problem constants ----------------
#define B_    8
#define L_    4096
#define H_    16
#define NB_   8
#define D_    1024
#define DH_   128
#define C_    128                 // H*NB
#define NTOK  (B_*L_)             // 32768
#define NTOT  ((size_t)NTOK*C_)   // 4194304

// scan chunking
#define CHUNKS 64
#define CLEN   (L_/CHUNKS)        // 64

// ---------------- workspace layout (bytes) ----------------
#define W1_FRAGS   (32*8)                       // K=1024/32 ksteps * 8 ntiles
#define W2_FRAGS   (4*8)
#define WS_W1R     0u
#define WS_W2R     (WS_W1R + W1_FRAGS*512u*2u)  // 262144
#define WS_W1M     (WS_W2R + W2_FRAGS*512u*2u)
#define WS_W2M     (WS_W1M + W1_FRAGS*512u*2u)
#define WS_RAWR    (1u<<20)
#define WS_RAWM    (WS_RAWR + (unsigned)(NTOT*4))
#define WS_COMPA   (WS_RAWM + (unsigned)(NTOT*4))
#define WS_COMPU   (WS_COMPA + B_*CHUNKS*C_*4u)
#define WS_CARRY   (WS_COMPU + B_*CHUNKS*C_*4u)

typedef __attribute__((ext_vector_type(16))) __bf16        v16bf;
typedef __attribute__((ext_vector_type(8)))  float         v8f;
typedef __attribute__((ext_vector_type(4)))  unsigned int  u32x4;
typedef __attribute__((ext_vector_type(8)))  int           i32x8;
typedef __attribute__((ext_vector_type(4)))  int           i32x4;

// ============================================================
// TDM helper: 1D copy of `nelem` 8-byte elements, global -> LDS.
// D# per CDNA5 ISA:
//  group0: [1:0]=count=1 | lds_addr | global_addr[56:0] | type=2
//  group1: data_size=3 (8B); tensor nelem x 1; tile nelem x 1
// This toolchain's builtin arity (probe-confirmed by the compile
// error): (u32x4, i32x8, i32x4, i32x4, i32x8, i32 cpol).
// ============================================================
__device__ __forceinline__ void tdm_load_1d(const void* gptr, unsigned lds_off,
                                            unsigned nelem) {
    unsigned long long ga = (unsigned long long)(uintptr_t)gptr;
    u32x4 g0 = { 1u,                                   // count=1 (user descriptor)
                 lds_off,                              // lds_addr (bytes)
                 (unsigned)(ga & 0xFFFFFFFFu),         // global_addr[31:0]
                 (unsigned)((ga >> 32) & 0x1FFFFFFu) | (2u << 30) }; // [56:32]|type=2
    i32x8 g1 = { (int)(3u << 16),                      // data_size = 8B
                 (int)((nelem & 0xFFFFu) << 16),       // tensor_dim0[15:0]
                 (int)(((nelem >> 16) & 0xFFFFu) | (1u << 16)), // td0[31:16]|td1=1
                 (int)((nelem & 0xFFFFu) << 16),       // tile_dim0
                 1,                                    // tile_dim1=1, tile_dim2=0
                 (int)nelem,                           // tensor_dim0_stride[31:0]
                 0, 0 };
    i32x4 z4 = { 0, 0, 0, 0 };
    i32x8 z8 = { 0, 0, 0, 0, 0, 0, 0, 0 };
    __builtin_amdgcn_tensor_load_to_lds(g0, g1, z4, z4, z8, 0);
}

__device__ __forceinline__ unsigned lds_offset_of(const void* p) {
    return (unsigned)(uintptr_t)p;   // generic LDS addr: low 32 bits = LDS byte offset
}

// ============================================================
// Kernel 0: repack fp32 row-major weight [K x 128] into bf16
// B-fragment order for V_WMMA_F32_16X16X32_BF16.
// 16-bit B (32x16): lanes 0-15 hold K=0..7,16..23, lanes 16-31 hold
// K=8..15,24..31; half h holds K = koff + h + (h & 8), koff = 0 / 8.
// Per-lane 16 halves contiguous (32B) -> fragment = 1KB.
// ============================================================
__global__ void repack_bfrag(const float* __restrict__ W,
                             unsigned short* __restrict__ dst, int ksteps) {
    int t = blockIdx.x * blockDim.x + threadIdx.x;
    int nfrag = ksteps * 8;
    if (t >= nfrag * 32) return;
    int frag = t >> 5, lane = t & 31;
    int kt = frag >> 3, nt = frag & 7;
    int kbase = kt * 32 + ((lane & 16) ? 8 : 0);
    int n = nt * 16 + (lane & 15);
    __bf16* d = (__bf16*)dst;
#pragma unroll
    for (int h = 0; h < 16; ++h)
        d[(size_t)frag * 512 + lane * 16 + h] = (__bf16)W[(size_t)(kbase + h + (h & 8)) * C_ + n];
}

// ============================================================
// Kernel 1: fused 2-layer MLP per head, bf16 WMMA, f32 accum.
// One wave owns 16 tokens; block = 8 waves = 128 tokens.
// Layer-1 weights are TDM-staged into double-buffered LDS
// (wave 0 issues tensor_load_to_lds for kt+2 while the block
// computes kt); B fragments then come from ds_load_b128.
// ============================================================
__global__ void __launch_bounds__(256)
mlp_head(const float* __restrict__ X,
         const unsigned short* __restrict__ w1p,
         const unsigned short* __restrict__ w2p,
         const float* __restrict__ b1,
         const float* __restrict__ b2,
         float* __restrict__ raw) {
    __shared__ __bf16 hid[8][16][136];     // per-wave 16x128 hidden, padded rows
    __shared__ __bf16 wbuf[2][4096];       // double-buffered 8KB weight stage

    const int wave = threadIdx.x >> 5;
    const int lane = threadIdx.x & 31;
    const int lmod = lane & 15;
    const int koff = (lane & 16) ? 8 : 0;  // A-fragment K sub-offset
    const int moff = (lane & 16) ? 8 : 0;  // C-layout row offset
    const int tb = (blockIdx.x * 8 + wave) * 16;
    const bool issuer = (threadIdx.x < 32);

    // prologue: TDM-prime both buffers (kt=0, kt=1); in-order completion
    if (issuer) {
        tdm_load_1d(w1p,          lds_offset_of(&wbuf[0][0]), 1024);
        tdm_load_1d(w1p + 4096,   lds_offset_of(&wbuf[1][0]), 1024);
    }

    v8f acc[8];
#pragma unroll
    for (int i = 0; i < 8; ++i) acc[i] = v8f{};

    const float* rowp = X + (size_t)(tb + lmod) * D_;

    for (int kt = 0; kt < 32; ++kt) {
        if (issuer) {
            if (kt == 31) __builtin_amdgcn_s_wait_tensorcnt(0);  // last tile: drain
            else          __builtin_amdgcn_s_wait_tensorcnt(1);  // tile kt done, kt+1 in flight
        }
        __syncthreads();                    // publish wbuf[kt&1] to all waves

        // A fragment: 16-bit A 16x32; halves 0-7 = K[kt*32+koff..+8), 8-15 = +16
        const float* p = rowp + kt * 32 + koff;
        float4 f0 = *(const float4*)(p);
        float4 f1 = *(const float4*)(p + 4);
        float4 f2 = *(const float4*)(p + 16);
        float4 f3 = *(const float4*)(p + 20);
        __builtin_prefetch(p + 32, 0, 3);   // near-scope prefetch of next K slice
        v16bf a;
        a[0]=(__bf16)f0.x;  a[1]=(__bf16)f0.y;  a[2]=(__bf16)f0.z;  a[3]=(__bf16)f0.w;
        a[4]=(__bf16)f1.x;  a[5]=(__bf16)f1.y;  a[6]=(__bf16)f1.z;  a[7]=(__bf16)f1.w;
        a[8]=(__bf16)f2.x;  a[9]=(__bf16)f2.y;  a[10]=(__bf16)f2.z; a[11]=(__bf16)f2.w;
        a[12]=(__bf16)f3.x; a[13]=(__bf16)f3.y; a[14]=(__bf16)f3.z; a[15]=(__bf16)f3.w;

        const __bf16* wb = &wbuf[kt & 1][lane * 16];
#pragma unroll
        for (int nt = 0; nt < 8; ++nt) {
            v16bf b = *(const v16bf*)(wb + nt * 512);
            acc[nt] = __builtin_amdgcn_wmma_f32_16x16x32_bf16(
                false, a, false, b, (short)0, acc[nt], false, false);
        }

        __syncthreads();                    // all waves done reading wbuf[kt&1]
        if (issuer && kt + 2 < 32)
            tdm_load_1d(w1p + (size_t)(kt + 2) * 4096,
                        lds_offset_of(&wbuf[kt & 1][0]), 1024);
    }

    // bias + exact GELU, transpose through LDS (wave-private region)
#pragma unroll
    for (int nt = 0; nt < 8; ++nt) {
        float bv = b1[nt * 16 + lmod];
#pragma unroll
        for (int r = 0; r < 8; ++r) {
            float x = acc[nt][r] + bv;
            float g = 0.5f * x * (1.0f + erff(x * 0.70710678118654752f));
            hid[wave][r + moff][nt * 16 + lmod] = (__bf16)g;
        }
    }

    v8f acc2[8];
#pragma unroll
    for (int i = 0; i < 8; ++i) acc2[i] = v8f{};

#pragma unroll
    for (int kt = 0; kt < 4; ++kt) {
        const __bf16* hrow = &hid[wave][lmod][kt * 32 + koff];
        v16bf a;
#pragma unroll
        for (int j = 0; j < 8; ++j) { a[j] = hrow[j]; a[8 + j] = hrow[16 + j]; }
        const unsigned short* wk = w2p + (size_t)kt * 8 * 512 + lane * 16;
#pragma unroll
        for (int nt = 0; nt < 8; ++nt) {
            v16bf b = *(const v16bf*)(wk + nt * 512);
            acc2[nt] = __builtin_amdgcn_wmma_f32_16x16x32_bf16(
                false, a, false, b, (short)0, acc2[nt], false, false);
        }
    }

    // epilogue: + b2, store raw f32 (token-major, channel-minor = theta layout)
#pragma unroll
    for (int nt = 0; nt < 8; ++nt) {
        float bv = b2[nt * 16 + lmod];
#pragma unroll
        for (int r = 0; r < 8; ++r) {
            int tok = tb + r + moff;
            raw[(size_t)tok * C_ + nt * 16 + lmod] = acc2[nt][r] + bv;
        }
    }
}

// ============================================================
// Kernel 2: fused elementwise. Produces Pi/K/R outputs and
// rewrites workspace in place with (alpha, u) for the scan.
// ============================================================
__global__ void fuse_ew(const float* __restrict__ theta,
                        const float* __restrict__ logQ,
                        const float* __restrict__ logRavg,
                        float* __restrict__ rawR,   // in: MLP-R raw ; out: alpha
                        float* __restrict__ rawM,   // in: MLP-M raw ; out: u
                        float* __restrict__ out) {
    size_t idx = (size_t)blockIdx.x * blockDim.x + threadIdx.x;
    if (idx >= NTOT) return;
    int c = (int)(idx & (C_ - 1));

    float Q    = expf(logQ[c]);
    float Ravg = expf(logRavg[c]);
    float Pi   = 0.5f * (-Q + sqrtf(Q * Q + 4.0f * Q * Ravg));

    float lr = rawR[idx];
    lr = fminf(fmaxf(lr, -5.0f), 5.0f);
    float R = expf(lr);

    float z    = 3.14159265358979323846f * tanhf(rawM[idx]);
    float diff = z - theta[idx];
    float nu   = atan2f(sinf(diff), cosf(diff));   // wrap to (-pi, pi]

    float K = Pi / fmaxf(Pi + R, 1e-8f);

    out[NTOT + idx]     = Pi;
    out[2 * NTOT + idx] = K;
    out[3 * NTOT + idx] = R;
    rawR[idx] = 1.0f - K;  // alpha
    rawM[idx] = K * nu;    // u
}

// ============================================================
// Kernel 3a/3b/3c: chunked affine scan over L per (b, h, nb).
// One thread per channel c -> 512B coalesced lines per step.
// ============================================================
__global__ void scan_chunks(const float* __restrict__ alpha,
                            const float* __restrict__ u,
                            float* __restrict__ compA,
                            float* __restrict__ compU) {
    int b = blockIdx.x >> 6, ch = blockIdx.x & 63, c = threadIdx.x;
    float A = 1.0f, U = 0.0f;
    size_t base = (((size_t)b * L_ + (size_t)ch * CLEN) << 7) + c;
    for (int t = 0; t < CLEN; ++t) {
        size_t i = base + ((size_t)t << 7);
        float a = alpha[i], uu = u[i];
        U = a * U + uu;
        A = a * A;
    }
    int pos = (b * CHUNKS + ch) * C_ + c;
    compA[pos] = A;
    compU[pos] = U;
}

__global__ void scan_carry(const float* __restrict__ compA,
                           const float* __restrict__ compU,
                           float* __restrict__ carry) {
    int t = blockIdx.x * blockDim.x + threadIdx.x;
    if (t >= B_ * C_) return;
    int b = t >> 7, c = t & 127;
    float PA = 1.0f, PU = 0.0f;
    for (int ch = 0; ch < CHUNKS; ++ch) {
        int pos = (b * CHUNKS + ch) * C_ + c;
        carry[pos] = PU;                    // d entering this chunk (d_{-1} = 0)
        float A = compA[pos], U = compU[pos];
        PU = A * PU + U;
        PA = A * PA;
    }
}

__global__ void scan_apply(const float* __restrict__ alpha,
                           const float* __restrict__ u,
                           const float* __restrict__ carry,
                           const float* __restrict__ theta,
                           float* __restrict__ out0) {
    int b = blockIdx.x >> 6, ch = blockIdx.x & 63, c = threadIdx.x;
    float d = carry[(b * CHUNKS + ch) * C_ + c];
    size_t base = (((size_t)b * L_ + (size_t)ch * CLEN) << 7) + c;
    for (int t = 0; t < CLEN; ++t) {
        size_t i = base + ((size_t)t << 7);
        d = alpha[i] * d + u[i];
        out0[i] = theta[i] + d;
    }
}

// ============================================================
extern "C" void kernel_launch(void* const* d_in, const int* in_sizes, int n_in,
                              void* d_out, int out_size, void* d_ws, size_t ws_size,
                              hipStream_t stream) {
    const float* theta   = (const float*)d_in[0];
    const float* x       = (const float*)d_in[1];
    const float* logQ    = (const float*)d_in[2];
    const float* logRavg = (const float*)d_in[3];
    const float* r_w1 = (const float*)d_in[4];
    const float* r_b1 = (const float*)d_in[5];
    const float* r_w2 = (const float*)d_in[6];
    const float* r_b2 = (const float*)d_in[7];
    const float* m_w1 = (const float*)d_in[8];
    const float* m_b1 = (const float*)d_in[9];
    const float* m_w2 = (const float*)d_in[10];
    const float* m_b2 = (const float*)d_in[11];

    char* ws = (char*)d_ws;
    unsigned short* w1r = (unsigned short*)(ws + WS_W1R);
    unsigned short* w2r = (unsigned short*)(ws + WS_W2R);
    unsigned short* w1m = (unsigned short*)(ws + WS_W1M);
    unsigned short* w2m = (unsigned short*)(ws + WS_W2M);
    float* rawR  = (float*)(ws + WS_RAWR);
    float* rawM  = (float*)(ws + WS_RAWM);
    float* compA = (float*)(ws + WS_COMPA);
    float* compU = (float*)(ws + WS_COMPU);
    float* carry = (float*)(ws + WS_CARRY);
    float* out   = (float*)d_out;

    // 0) weight repack to bf16 WMMA B-fragment order
    repack_bfrag<<<(W1_FRAGS * 32 + 255) / 256, 256, 0, stream>>>(r_w1, w1r, 32);
    repack_bfrag<<<(W2_FRAGS * 32 + 255) / 256, 256, 0, stream>>>(r_w2, w2r, 4);
    repack_bfrag<<<(W1_FRAGS * 32 + 255) / 256, 256, 0, stream>>>(m_w1, w1m, 32);
    repack_bfrag<<<(W2_FRAGS * 32 + 255) / 256, 256, 0, stream>>>(m_w2, w2m, 4);

    // 1) fused MLP per head (32768 tokens / 128 per block)
    mlp_head<<<NTOK / 128, 256, 0, stream>>>(x, w1r, w2r, r_b1, r_b2, rawR);
    mlp_head<<<NTOK / 128, 256, 0, stream>>>(x, w1m, w2m, m_b1, m_b2, rawM);

    // 2) elementwise fuse: Pi/K/R out, alpha/u into workspace (in place)
    fuse_ew<<<(unsigned)((NTOT + 255) / 256), 256, 0, stream>>>(
        theta, logQ, logRavg, rawR, rawM, out);

    // 3) chunked affine scan over sequence dim
    scan_chunks<<<B_ * CHUNKS, C_, 0, stream>>>(rawR, rawM, compA, compU);
    scan_carry<<<(B_ * C_ + 255) / 256, 256, 0, stream>>>(compA, compU, carry);
    scan_apply<<<B_ * CHUNKS, C_, 0, stream>>>(rawR, rawM, carry, theta, out);
}
